// GraphProbeFeatures_35673998360737
// MI455X (gfx1250) — compile-verified, matrix operand order checked by reading, not run on hardware
//
#include <hip/hip_runtime.h>
#include <hip/hip_bf16.h>

typedef __attribute__((ext_vector_type(16))) _Float16 v16h;
typedef __attribute__((ext_vector_type(8)))  float    v8f;

#define BATCH 128
#define NPTS  4096
#define HID   128
#define PROJ  128
#define NT    128        // points per tile
#define NTILE (NPTS / NT)
#define TPB   256        // 8 waves
#define W0S   30.0f
#define LNEPS 1e-5f
#define LDH   136        // padded f16 row stride for activation tiles
#define TILEH 16384      // halfs per swizzled pw tile (32KB)

union Frag  { v16h h; uint4 q[2]; };
union Pack2 { _Float16 e[2]; uint32_t u; };
union Pack8 { _Float16 e[8]; uint4 q; };

// B-matrix swizzle: element (k in [0,128), n in [0,128)) -> fragment-order slot.
static __device__ __forceinline__ int bswz(int k, int n) {
  int kk = k >> 5, j = n >> 4, kl = k & 31;
  int lane = (n & 15) + (((kl >> 3) & 1) << 4);
  int slot = (kl & 7) + (((kl >> 4) & 1) << 3);
  return (((kk << 3) + j) * 32 + lane) * 16 + slot;
}

// A fragment (16x32 f16, M rows x K), row-major source with stride LDH halfs.
static __device__ __forceinline__ Frag ldA(const _Float16* buf, int row0, int kOff, int lane) {
  Frag f;
  int m = row0 + (lane & 15);
  int hh = lane >> 4;
  const _Float16* p0 = buf + m * LDH + kOff + hh * 8;
  f.q[0] = *(const uint4*)(p0);
  f.q[1] = *(const uint4*)(p0 + 16);
  return f;
}

// B fragment (32x16 f16) from a pre-swizzled buffer: 16 contiguous halfs per lane.
static __device__ __forceinline__ Frag ldB(const _Float16* buf, int kk, int j, int lane) {
  Frag f;
  const uint4* p = (const uint4*)(buf + ((((kk << 3) + j) * 32 + lane) << 4));
  f.q[0] = p[0];
  f.q[1] = p[1];
  return f;
}

// 128x128x128 GEMM step: preload all B fragments per K-chunk so the compiler can
// issue 16 ds_load_b128 back-to-back and stagger the s_wait_dscnt before WMMAs.
static __device__ __forceinline__ void gemm128(const _Float16* Abuf, const _Float16* Bbuf,
                                               v8f (&acc)[8], int wave, int lane) {
#pragma unroll
  for (int kk = 0; kk < 4; ++kk) {
    Frag a = ldA(Abuf, wave * 16, kk * 32, lane);
    Frag bf[8];
#pragma unroll
    for (int j = 0; j < 8; ++j) bf[j] = ldB(Bbuf, kk, j, lane);
#pragma unroll
    for (int j = 0; j < 8; ++j)
      acc[j] = __builtin_amdgcn_wmma_f32_16x16x32_f16(
          false, a.h, false, bf[j].h, (short)0, acc[j], false, false);
  }
}

// Async-DMA one 32KB tile from global (pre-swizzled f16) into LDS.
// 256 threads x 16B = 4KB per issue; 8 issues with matching global/LDS offsets.
static __device__ __forceinline__ void async_copy_tile(_Float16* ldsbuf,
                                                       const _Float16* gsrc, int tid) {
  uint32_t lda = (uint32_t)(uintptr_t)ldsbuf + ((uint32_t)tid << 4);
  uint64_t ga  = (uint64_t)(uintptr_t)gsrc + ((uint64_t)tid << 4);
  asm volatile(
      "global_load_async_to_lds_b128 %0, %1, off\n\t"
      "global_load_async_to_lds_b128 %0, %1, off offset:4096\n\t"
      "global_load_async_to_lds_b128 %0, %1, off offset:8192\n\t"
      "global_load_async_to_lds_b128 %0, %1, off offset:12288\n\t"
      "global_load_async_to_lds_b128 %0, %1, off offset:16384\n\t"
      "global_load_async_to_lds_b128 %0, %1, off offset:20480\n\t"
      "global_load_async_to_lds_b128 %0, %1, off offset:24576\n\t"
      "global_load_async_to_lds_b128 %0, %1, off offset:28672"
      :: "v"(lda), "v"(ga) : "memory");
}

static __device__ __forceinline__ void wait_async_all() {
  asm volatile("s_wait_asynccnt 0x0" ::: "memory");
}

// LayerNorm over P=128 from register fragments + store to global.
static __device__ __forceinline__ void ln_store(v8f (&z)[8], int layerIdx, int rowOff,
                                                int b, int wave, int lane,
                                                const float* pbg, const float* lngg,
                                                const float* lnbg, float* out) {
  int pl = lane & 15, hi = lane >> 4;
  float gv[8], bv[8], pv[8];
#pragma unroll
  for (int j = 0; j < 8; ++j) {
    int p = 16 * j + pl;
    gv[j] = lngg[layerIdx * 128 + p];
    bv[j] = lnbg[layerIdx * 128 + p];
    pv[j] = pbg[layerIdx * 128 + p];
  }
#pragma unroll
  for (int j = 0; j < 8; ++j)
#pragma unroll
    for (int r = 0; r < 8; ++r) z[j][r] += pv[j];

#pragma unroll
  for (int r = 0; r < 8; ++r) {
    float s = 0.f;
#pragma unroll
    for (int j = 0; j < 8; ++j) s += z[j][r];
    s += __shfl_xor(s, 1, 16); s += __shfl_xor(s, 2, 16);
    s += __shfl_xor(s, 4, 16); s += __shfl_xor(s, 8, 16);
    float mean = s * (1.0f / 128.0f);
    float v = 0.f;
#pragma unroll
    for (int j = 0; j < 8; ++j) { float d = z[j][r] - mean; v += d * d; }
    v += __shfl_xor(v, 1, 16); v += __shfl_xor(v, 2, 16);
    v += __shfl_xor(v, 4, 16); v += __shfl_xor(v, 8, 16);
    float rs = rsqrtf(v * (1.0f / 128.0f) + LNEPS);
    int ch = 16 * wave + r + 8 * hi;
    size_t rowbase = ((size_t)b * 261 + rowOff + ch) * 128;
#pragma unroll
    for (int j = 0; j < 8; ++j)
      out[rowbase + 16 * j + pl] = (z[j][r] - mean) * rs * gv[j] + bv[j];
  }
}

// One-time prep: convert pw layers 1,2 (f32) into fragment-order f16 tiles in d_ws.
// Tile t = layer*NTILE + nt holds TILEH halfs. 64 blocks, one tile each.
__global__ __launch_bounds__(TPB)
void GraphProbeFeatures_prep_kernel(const float* __restrict__ pwg,
                                    _Float16* __restrict__ wsh) {
  const int l  = blockIdx.x >> 5;        // 0 -> pw1, 1 -> pw2
  const int nt = blockIdx.x & 31;
  const float* src = pwg + (size_t)(1 + l) * NPTS * PROJ + (size_t)nt * NT * PROJ;
  _Float16* dst = wsh + (size_t)blockIdx.x * TILEH;
#pragma unroll 4
  for (int s = 0; s < TILEH / TPB; ++s) {
    int e = s * TPB + threadIdx.x;
    int k = e >> 7, p = e & 127;
    dst[bswz(k, p)] = (_Float16)src[(size_t)k * PROJ + p];
  }
}

__global__ __launch_bounds__(TPB, 1)
void GraphProbeFeatures_35673998360737_kernel(
    const float* __restrict__ w0g, const float* __restrict__ w1g,
    const float* __restrict__ w2g, const float* __restrict__ b0g,
    const float* __restrict__ b1g, const float* __restrict__ b2g,
    const float* __restrict__ xing, const float* __restrict__ pwg,
    const float* __restrict__ pbg, const float* __restrict__ lngg,
    const float* __restrict__ lnbg, const _Float16* __restrict__ wsh,
    float* __restrict__ out) {
  __shared__ _Float16 sH1[NT * LDH];     // h1 tile, [n][ch]
  __shared__ _Float16 sHT[HID * LDH];    // h1ᵀ then h2ᵀ tile, [ch][n]
  __shared__ _Float16 sW1[HID * HID];    // W1, swizzled B fragments
  __shared__ _Float16 sPW[TILEH];        // pw tile, swizzled (async-staged)
  __shared__ float sX[NT * 2];
  __shared__ float sW0[2 * HID];
  __shared__ float sB0[HID];
  __shared__ float sB1[HID];
  __shared__ float sW2[HID * 4];
  __shared__ float sB2[4];
  __shared__ float sH3[NT * 4];
  __shared__ float sZ[5 * 128];
  __shared__ float sStat[10];

  const int b    = blockIdx.x;
  const int tid  = threadIdx.x;
  const int wave = tid >> 5;
  const int lane = tid & 31;

  const float* pw0g = pwg;
  const float* pw3g = pwg + (size_t)3 * NPTS * PROJ;

  // ---- one-time per-sample setup ----
  sW0[tid] = w0g[(size_t)b * 2 * HID + tid];                // [2][128]
  if (tid < HID) { sB0[tid] = b0g[b * HID + tid]; sB1[tid] = b1g[b * HID + tid]; }
  if (tid < 3) sB2[tid] = b2g[b * 3 + tid];
  for (int e = tid; e < HID * 3; e += TPB) {
    int k = e / 3, d = e - 3 * k;
    sW2[k * 4 + d] = w2g[((size_t)b * HID + k) * 3 + d];
  }
#pragma unroll 4
  for (int s = 0; s < 64; ++s) {
    int e = s * TPB + tid;
    int k = e >> 7, o = e & 127;
    sW1[bswz(k, o)] = (_Float16)w1g[((size_t)b * HID + k) * HID + o];
  }

  // persistent projection accumulators: wave owns channels [16*wave, 16*wave+16)
  v8f zacc1[8], zacc2[8];
#pragma unroll
  for (int j = 0; j < 8; ++j)
#pragma unroll
    for (int r = 0; r < 8; ++r) { zacc1[j][r] = 0.f; zacc2[j][r] = 0.f; }

  float acc0 = 0.f, acc3a = 0.f, acc3b = 0.f;
  const int p0i = tid & 127, d0i = tid >> 7;

  for (int nt = 0; nt < NTILE; ++nt) {
    const int n0 = nt * NT;
    __syncthreads();   // previous iteration's readers of sPW/sX/sH3/sHT are done

    // stage pw1 tile (async DMA, overlaps Phase A) + load coords tile
    async_copy_tile(sPW, wsh + (size_t)nt * TILEH, tid);
    sX[tid] = xing[(size_t)n0 * 2 + tid];
    __syncthreads();

    // Phase A: h1 = sin(30*(x@W0+b0)) -> sH1 [n][ch] and sHT [ch][n]
    {
      int n = tid & 127;
      int cb = (tid >> 7) << 6;  // 0 or 64
      float x0 = sX[n * 2 + 0], x1 = sX[n * 2 + 1];
#pragma unroll 8
      for (int i = 0; i < 64; i += 2) {
        int ch = cb + i;
        float v0 = __sinf(W0S * (x0 * sW0[ch]     + x1 * sW0[HID + ch]     + sB0[ch]));
        float v1 = __sinf(W0S * (x0 * sW0[ch + 1] + x1 * sW0[HID + ch + 1] + sB0[ch + 1]));
        Pack2 pk; pk.e[0] = (_Float16)v0; pk.e[1] = (_Float16)v1;
        *(uint32_t*)(sH1 + n * LDH + ch) = pk.u;
        sHT[ch * LDH + n]       = pk.e[0];
        sHT[(ch + 1) * LDH + n] = pk.e[1];
      }
    }
    wait_async_all();     // pw1 tile landed
    __syncthreads();

    // Phase P1: zacc1 += h1ᵀ @ pw1_tile  (M=ch strip, K=n, N=P)
    gemm128(sHT, sPW, zacc1, wave, lane);
    __syncthreads();      // all waves done reading sPW and sHT

    // stage pw2 tile (async DMA, overlaps Phase B)
    async_copy_tile(sPW, wsh + (size_t)(NTILE + nt) * TILEH, tid);

    // Phase B: h2 = sin(30*(h1@W1+b1)) -> sHT (transposed)
    {
      v8f cacc[8];
#pragma unroll
      for (int j = 0; j < 8; ++j)
#pragma unroll
        for (int r = 0; r < 8; ++r) cacc[j][r] = 0.f;
      gemm128(sH1, sW1, cacc, wave, lane);
      int hi = lane >> 4;
#pragma unroll
      for (int j = 0; j < 8; ++j) {
        int ch = 16 * j + (lane & 15);
        float bb = sB1[ch];
        Pack8 pk;
#pragma unroll
        for (int r = 0; r < 8; ++r)
          pk.e[r] = (_Float16)__sinf(W0S * (cacc[j][r] + bb));
        *(uint4*)(sHT + ch * LDH + 16 * wave + 8 * hi) = pk.q;
      }
    }
    __syncthreads();

    // h3 tile (VALU, reads sHT = h2ᵀ); prefetch next tile's pw0/pw3 rows
    for (int e = tid; e < NT * 3; e += TPB) {
      int n = e / 3, d = e - 3 * n;
      float acc = sB2[d];
#pragma unroll 8
      for (int k = 0; k < HID; ++k)
        acc += (float)sHT[k * LDH + n] * sW2[k * 4 + d];
      sH3[n * 4 + d] = acc;
    }
    if (nt + 1 < NTILE) {
      __builtin_prefetch(pw0g + (size_t)(n0 + NT) * PROJ + p0i, 0, 0);
      __builtin_prefetch(pw3g + (size_t)(n0 + NT) * PROJ + p0i, 0, 0);
    }
    wait_async_all();     // pw2 tile landed
    __syncthreads();

    // Phase P2: zacc2 += h2ᵀ @ pw2_tile
    gemm128(sHT, sPW, zacc2, wave, lane);

    // Phase Z: tiny layer-0 and layer-3 projections (VALU, coalesced global reads)
    {
      const float* pwr0 = pw0g + (size_t)n0 * PROJ + p0i;
      const float* pwr3 = pw3g + (size_t)n0 * PROJ + p0i;
      float a0 = 0.f, a3 = 0.f;
#pragma unroll 4
      for (int n = 0; n < NT; ++n) {
        float pwv0 = pwr0[(size_t)n * PROJ];
        float pwv3 = pwr3[(size_t)n * PROJ];
        a0 += sX[n * 2 + d0i] * pwv0;
        a3 += sH3[n * 4 + d0i] * pwv3;
      }
      acc0 += a0; acc3a += a3;
      if (tid < 128) {
        const float* pwr = pw3g + (size_t)n0 * PROJ + tid;
        float a3b = 0.f;
#pragma unroll 4
        for (int n = 0; n < NT; ++n)
          a3b += sH3[n * 4 + 2] * pwr[(size_t)n * PROJ];
        acc3b += a3b;
      }
    }
  }

  // ---- epilogues: LayerNorm + store ----
  ln_store(zacc1, 1, 2,   b, wave, lane, pbg, lngg, lnbg, out);   // rows 2..129
  ln_store(zacc2, 2, 130, b, wave, lane, pbg, lngg, lnbg, out);   // rows 130..257

  __syncthreads();
  sZ[d0i * 128 + p0i]       = acc0  + pbg[p0i];                   // layer0 rows 0..1
  sZ[(2 + d0i) * 128 + p0i] = acc3a + pbg[3 * 128 + p0i];         // layer3 d=0,1
  if (tid < 128) sZ[4 * 128 + tid] = acc3b + pbg[3 * 128 + tid];  // layer3 d=2
  __syncthreads();
  if (tid < 5) {
    const float* row = sZ + tid * 128;
    float s = 0.f;
    for (int p = 0; p < 128; ++p) s += row[p];
    float mean = s * (1.0f / 128.0f);
    float v = 0.f;
    for (int p = 0; p < 128; ++p) { float d = row[p] - mean; v += d * d; }
    sStat[2 * tid]     = mean;
    sStat[2 * tid + 1] = rsqrtf(v * (1.0f / 128.0f) + LNEPS);
  }
  __syncthreads();
  {
    float m = sStat[2 * d0i], rs = sStat[2 * d0i + 1];
    out[((size_t)b * 261 + d0i) * 128 + p0i] =
        (sZ[d0i * 128 + p0i] - m) * rs * lngg[p0i] + lnbg[p0i];
    int rr = 2 + d0i;
    m = sStat[2 * rr]; rs = sStat[2 * rr + 1];
    out[((size_t)b * 261 + 258 + d0i) * 128 + p0i] =
        (sZ[rr * 128 + p0i] - m) * rs * lngg[3 * 128 + p0i] + lnbg[3 * 128 + p0i];
    if (tid < 128) {
      m = sStat[8]; rs = sStat[9];
      out[((size_t)b * 261 + 260) * 128 + tid] =
          (sZ[4 * 128 + tid] - m) * rs * lngg[3 * 128 + tid] + lnbg[3 * 128 + tid];
    }
  }
}

extern "C" void kernel_launch(void* const* d_in, const int* in_sizes, int n_in,
                              void* d_out, int out_size, void* d_ws, size_t ws_size,
                              hipStream_t stream) {
  (void)in_sizes; (void)n_in; (void)out_size; (void)ws_size;
  const float* w0  = (const float*)d_in[0];
  const float* w1  = (const float*)d_in[1];
  const float* w2  = (const float*)d_in[2];
  const float* b0  = (const float*)d_in[3];
  const float* b1  = (const float*)d_in[4];
  const float* b2  = (const float*)d_in[5];
  const float* xin = (const float*)d_in[6];
  const float* pw  = (const float*)d_in[7];
  const float* pb  = (const float*)d_in[8];
  const float* lng = (const float*)d_in[9];
  const float* lnb = (const float*)d_in[10];
  float* out = (float*)d_out;
  _Float16* wsh = (_Float16*)d_ws;   // 2 MB: 64 swizzled f16 pw tiles

  GraphProbeFeatures_prep_kernel<<<dim3(2 * NTILE), dim3(TPB), 0, stream>>>(pw, wsh);
  GraphProbeFeatures_35673998360737_kernel<<<dim3(BATCH), dim3(TPB), 0, stream>>>(
      w0, w1, w2, b0, b1, b2, xin, pw, pb, lng, lnb, wsh, out);
}
